// Synth_Attention_24970939859055
// MI455X (gfx1250) — compile-verified, matrix operand order checked by reading, not run on hardware
//
#include <hip/hip_runtime.h>
#include <hip/hip_bf16.h>

// ---------------------------------------------------------------------------
// Sizes (fixed by the reference)
#define NH   8      // heads
#define DIN  512    // input dim
#define DE   64     // head dim
#define NB   8      // batch
#define NQ   1024   // sequence
// ---------------------------------------------------------------------------

typedef __bf16 bf16_t;
typedef __attribute__((ext_vector_type(16))) __bf16 v16bf;
typedef __attribute__((ext_vector_type(8)))  float  v8f;
typedef __attribute__((ext_vector_type(4)))  unsigned int u32x4;

union FragU { u32x4 q[2]; v16bf v; };
union F4U   { u32x4 q; float f[4]; };

// Load a 16-lane-striped bf16 WMMA fragment: each lane supplies two 16-byte
// contiguous chunks (VGPR0-3 chunk, VGPR4-7 chunk) per the CDNA5 16-bit
// A/B layout (lane = row/col, 8 contiguous K per half).
static __device__ inline v16bf load_frag(const bf16_t* p0, const bf16_t* p1) {
    FragU f;
    f.q[0] = *(const u32x4*)p0;
    f.q[1] = *(const u32x4*)p1;
    return f.v;
}

static __device__ inline v8f wmma_bf16(v16bf a, v16bf b, v8f c) {
    return __builtin_amdgcn_wmma_f32_16x16x32_bf16(false, a, false, b,
                                                   (short)0, c, false, false);
}

static __device__ inline v8f zero8() {
    v8f z; for (int i = 0; i < 8; ++i) z[i] = 0.0f; return z;
}

// ---------------------------------------------------------------------------
// Prep: f32->bf16 convert h_fea; transpose-convert Wq/Wk/Wv -> [h][e][k]
// and W_out -> [d][h*64+e] so every WMMA B-fragment load is 16B contiguous.
// ---------------------------------------------------------------------------
__global__ void prep_kernel(const float* __restrict__ h_fea,
                            const float* __restrict__ Wq,
                            const float* __restrict__ Wk,
                            const float* __restrict__ Wv,
                            const float* __restrict__ Wo,
                            bf16_t* __restrict__ hbf,
                            bf16_t* __restrict__ Wqt,
                            bf16_t* __restrict__ Wkt,
                            bf16_t* __restrict__ Wvt,
                            bf16_t* __restrict__ Wot) {
    size_t tid    = (size_t)blockIdx.x * blockDim.x + threadIdx.x;
    size_t stride = (size_t)gridDim.x * blockDim.x;
    for (size_t i = tid; i < (size_t)NB * NQ * DIN; i += stride)
        hbf[i] = (bf16_t)h_fea[i];
    for (size_t i = tid; i < (size_t)NH * DE * DIN; i += stride) {
        // Wt layout [h][e][k] : i = (h*DE + e)*DIN + k ; src [h][k][e]
        size_t k = i & (DIN - 1);
        size_t he = i >> 9;             // DIN = 512
        size_t e = he & (DE - 1);
        size_t h = he >> 6;
        size_t src = (h * DIN + k) * DE + e;
        Wqt[i] = (bf16_t)Wq[src];
        Wkt[i] = (bf16_t)Wk[src];
        Wvt[i] = (bf16_t)Wv[src];
        // Wot layout [d][h2*64+e2] : i = d*512 + he2 ; src Wo[(h2*64+e2)*512 + d]
        size_t d   = i >> 9;
        size_t he2 = i & 511;
        Wot[i] = (bf16_t)Wo[he2 * DIN + d];
    }
}

// ---------------------------------------------------------------------------
// QKV projection:  [8192 rows (b,n)] x [512] x [64] per head, bf16 WMMA.
// grid = (64 row-groups, 8 heads), 8 waves/block, one 16-row tile per wave.
// Q,K stored row-major [h][b][n][64]; V stored transposed [h][b][e][n]
// so the attention P*V B-fragments are contiguous.
// ---------------------------------------------------------------------------
__global__ __launch_bounds__(256)
void qkv_kernel(const bf16_t* __restrict__ hbf,
                const bf16_t* __restrict__ Wqt,
                const bf16_t* __restrict__ Wkt,
                const bf16_t* __restrict__ Wvt,
                bf16_t* __restrict__ Qb,
                bf16_t* __restrict__ Kb,
                bf16_t* __restrict__ Vt) {
    const int h    = blockIdx.y;
    const int wave = threadIdx.x >> 5;
    const int lane = threadIdx.x & 31;
    const int li   = lane & 15;
    const int lh   = lane >> 4;
    const int koffs = lh * 8;
    const int rowtile = blockIdx.x * 8 + wave;   // 0..511
    const int r0 = rowtile * 16;

    v8f aq[4], ak[4], av[4];
    for (int e = 0; e < 4; ++e) { aq[e] = zero8(); ak[e] = zero8(); av[e] = zero8(); }

    for (int kb = 0; kb < DIN; kb += 32) {
        const bf16_t* arow = hbf + (size_t)(r0 + li) * DIN + kb + koffs;
        v16bf a = load_frag(arow, arow + 16);
        for (int et = 0; et < 4; ++et) {
            const size_t wo = ((size_t)h * DE + et * 16 + li) * DIN + kb + koffs;
            aq[et] = wmma_bf16(a, load_frag(Wqt + wo, Wqt + wo + 16), aq[et]);
            ak[et] = wmma_bf16(a, load_frag(Wkt + wo, Wkt + wo + 16), ak[et]);
            av[et] = wmma_bf16(a, load_frag(Wvt + wo, Wvt + wo + 16), av[et]);
        }
    }
    // C layout: VGPR v, lanes 0-15 -> M=v, lanes 16-31 -> M=v+8; N = li
    for (int et = 0; et < 4; ++et) {
        for (int v = 0; v < 8; ++v) {
            int gr = r0 + v + 8 * lh;           // flattened (b,n)
            int b  = gr >> 10;
            int n  = gr & (NQ - 1);
            int e  = et * 16 + li;
            size_t qi = (((size_t)h * NB + b) * NQ + n) * DE + e;
            Qb[qi] = (bf16_t)aq[et][v];
            Kb[qi] = (bf16_t)ak[et][v];
            size_t vi = (((size_t)h * NB + b) * DE + e) * NQ + n;
            Vt[vi] = (bf16_t)av[et][v];
        }
    }
}

// ---------------------------------------------------------------------------
// Fused attention: per workgroup = (batch b, 16-row n-tile), 8 waves = 8 heads.
// Streams m in 32-wide tiles: QK^T (WMMA) -> LDS compat -> 16->16->8 MLP
// (pointwise, 256 threads) -> per-head online softmax -> P*V (WMMA, K=32),
// then fused head-combine GEMM with W_out (WMMA, K=512).
// The 256MB aux stream (read + passthrough write) is issued NON-TEMPORAL so
// it does not evict the L2-resident K/V/Wout working set; a software
// prefetch (global_prefetch_b8) runs one tile ahead of the stream.
// ---------------------------------------------------------------------------
#define SM_COMPAT  (16 * 32 * 17 * 4)          // 34816 B, pad 17 kills bank conflicts
#define SM_SCORE   (NH * 16 * 32 * 4)          // 16384 B
#define SM_PTILE   (NH * 16 * 32 * 2)          //  8192 B
#define SM_COEF    (256 * 4 + 16 * 4 + 128 * 4 + 8 * 4 + 32)

__global__ __launch_bounds__(256)
void attn_kernel(const bf16_t* __restrict__ Qb,
                 const bf16_t* __restrict__ Kb,
                 const bf16_t* __restrict__ Vt,
                 const float* __restrict__ aux,
                 const float* __restrict__ W1, const float* __restrict__ b1,
                 const float* __restrict__ W2, const float* __restrict__ b2,
                 const bf16_t* __restrict__ Wot,
                 float* __restrict__ out_hwave,
                 float* __restrict__ out_aux) {
    __shared__ __align__(16) char smem[SM_COMPAT + SM_SCORE + SM_PTILE + SM_COEF];
    float  (*compat)[32][17] = (float (*)[32][17])smem;
    char*   dynp   = smem + SM_COMPAT;
    float  (*s_tile)[16][32] = (float (*)[16][32])dynp;
    bf16_t (*p_tile)[16][32] = (bf16_t (*)[16][32])(dynp + SM_SCORE);
    bf16_t (*hd)[DIN]        = (bf16_t (*)[DIN])dynp;   // aliases s_tile (post-loop)
    float* w1s = (float*)(smem + SM_COMPAT + SM_SCORE + SM_PTILE);
    float* b1s = w1s + 256;
    float* w2s = b1s + 16;
    float* b2s = w2s + 128;

    const int b  = blockIdx.x >> 6;
    const int n0 = (blockIdx.x & 63) * 16;
    const int h    = threadIdx.x >> 5;     // wave h owns head h
    const int lane = threadIdx.x & 31;
    const int li   = lane & 15;
    const int lh   = lane >> 4;
    const int koffs = lh * 8;
    const int t = threadIdx.x;

    w1s[t] = W1[t];
    if (t < 128) w2s[t] = W2[t];
    if (t < 16)  b1s[t] = b1[t];
    if (t < 8)   b2s[t] = b2[t];

    // Q fragments for this head: rows n0..n0+15, K = 0..63 (two 16x32 frags)
    const bf16_t* qbase = Qb + (((size_t)h * NB + b) * NQ + n0 + li) * DE + koffs;
    v16bf aq0 = load_frag(qbase,      qbase + 16);
    v16bf aq1 = load_frag(qbase + 32, qbase + 48);

    // per-wave aux stream base: row (n0+li), 16-float half selected by lh
    const size_t abase =
        (((size_t)h * NB + b) * NQ + (n0 + li)) * NQ + lh * 16;

    float rmax[8], rsum[8];
    v8f acc[4];
    for (int v = 0; v < 8; ++v) { rmax[v] = -3.0e38f; rsum[v] = 0.0f; }
    for (int e = 0; e < 4; ++e) acc[e] = zero8();

    __syncthreads();

    for (int m0 = 0; m0 < NQ; m0 += 32) {
        // (a) qk tile 16x32 for head h -> LDS compat[.][.][h]
        for (int ms = 0; ms < 2; ++ms) {
            v8f cq = zero8();
            const bf16_t* kbase =
                Kb + (((size_t)h * NB + b) * NQ + m0 + ms * 16 + li) * DE + koffs;
            cq = wmma_bf16(aq0, load_frag(kbase,      kbase + 16), cq);
            cq = wmma_bf16(aq1, load_frag(kbase + 32, kbase + 48), cq);
            for (int v = 0; v < 8; ++v)
                compat[v + 8 * lh][ms * 16 + li][h] = cq[v];
        }
        // (b) aux load -> compat feature 8+h, fused with passthrough store.
        //     Non-temporal (TH=NT) both ways: stream-once data must not evict
        //     the L2-resident K/V/Wout working set. Prefetch one tile ahead.
        {
            const size_t aoff = abase + m0;
            if (m0 + 32 < NQ)
                __builtin_prefetch(aux + aoff + 32, 0, 0);   // global_prefetch_b8
            const u32x4* ap = (const u32x4*)(aux + aoff);
            u32x4*       op = (u32x4*)(out_aux + aoff);
            float vals[16];
            for (int q = 0; q < 4; ++q) {
                F4U u;
                u.q = __builtin_nontemporal_load(ap + q);
                __builtin_nontemporal_store(u.q, op + q);
                vals[q * 4 + 0] = u.f[0]; vals[q * 4 + 1] = u.f[1];
                vals[q * 4 + 2] = u.f[2]; vals[q * 4 + 3] = u.f[3];
            }
            for (int j = 0; j < 16; ++j)
                compat[li][lh * 16 + j][8 + h] = vals[j];
        }
        __syncthreads();

        // (c) pointwise score-aggregation MLP over the 16x32 tile (512 elems)
        for (int e2 = 0; e2 < 2; ++e2) {
            int idx = e2 * 256 + t;
            int nn = idx >> 5, mm = idx & 31;
            float f[16], x[16];
            for (int j = 0; j < 16; ++j) f[j] = compat[nn][mm][j];
            for (int j = 0; j < 16; ++j) {
                float s = b1s[j];
                for (int k = 0; k < 16; ++k) s = fmaf(w1s[j * 16 + k], f[k], s);
                x[j] = fmaxf(s, 0.0f);
            }
            for (int h2 = 0; h2 < 8; ++h2) {
                float s = b2s[h2];
                for (int k = 0; k < 16; ++k) s = fmaf(w2s[h2 * 16 + k], x[k], s);
                s_tile[h2][nn][mm] = s;
            }
        }
        __syncthreads();

        // (d) wave h: online softmax over its head + P*V accumulation
        v8f sc0, sc1;
        for (int v = 0; v < 8; ++v) {
            sc0[v] = s_tile[h][v + 8 * lh][li];
            sc1[v] = s_tile[h][v + 8 * lh][16 + li];
        }
        float tmax[8];
        for (int v = 0; v < 8; ++v) tmax[v] = fmaxf(sc0[v], sc1[v]);
        for (int off = 1; off < 16; off <<= 1)
            for (int v = 0; v < 8; ++v)
                tmax[v] = fmaxf(tmax[v], __shfl_xor(tmax[v], off, 32));
        float psum[8];
        for (int v = 0; v < 8; ++v) {
            float nm   = fmaxf(rmax[v], tmax[v]);
            float corr = __expf(rmax[v] - nm);
            float p0   = __expf(sc0[v] - nm);
            float p1   = __expf(sc1[v] - nm);
            p_tile[h][v + 8 * lh][li]      = (bf16_t)p0;
            p_tile[h][v + 8 * lh][16 + li] = (bf16_t)p1;
            psum[v] = p0 + p1;
            rsum[v] = rsum[v] * corr;
            rmax[v] = nm;
            for (int e = 0; e < 4; ++e) acc[e][v] *= corr;
        }
        for (int off = 1; off < 16; off <<= 1)
            for (int v = 0; v < 8; ++v)
                psum[v] += __shfl_xor(psum[v], off, 32);
        for (int v = 0; v < 8; ++v) rsum[v] += psum[v];

        const bf16_t* pb = &p_tile[h][li][koffs];
        v16bf ap = load_frag(pb, pb + 16);          // P as A-frag (16x32, K=m)
        for (int et = 0; et < 4; ++et) {
            const bf16_t* vb =
                Vt + (((size_t)h * NB + b) * DE + et * 16 + li) * NQ + m0 + koffs;
            acc[et] = wmma_bf16(ap, load_frag(vb, vb + 16), acc[et]);
        }
        __syncthreads();
    }

    // normalize and stage heads tile [16 n][h*64+e] in bf16 (aliases score LDS)
    for (int v = 0; v < 8; ++v) {
        float inv = 1.0f / rsum[v];
        for (int et = 0; et < 4; ++et)
            hd[v + 8 * lh][h * DE + et * 16 + li] = (bf16_t)(acc[et][v] * inv);
    }
    __syncthreads();

    // fused output projection: [16 x 512] = heads[16 x 512] @ Wout[512 x 512]
    // wave h covers 4 of the 32 d-tiles. Output is write-once -> NT stores.
    for (int i = 0; i < 4; ++i) {
        const int d0 = (h * 4 + i) * 16;
        v8f co = zero8();
        for (int kb = 0; kb < DIN; kb += 32) {
            const bf16_t* ah = &hd[li][kb + koffs];
            const bf16_t* bw = Wot + (size_t)(d0 + li) * DIN + kb + koffs;
            co = wmma_bf16(load_frag(ah, ah + 16), load_frag(bw, bw + 16), co);
        }
        for (int v = 0; v < 8; ++v) {
            int n = n0 + v + 8 * lh;
            __builtin_nontemporal_store(
                co[v], &out_hwave[((size_t)b * NQ + n) * DIN + d0 + li]);
        }
    }
}

// ---------------------------------------------------------------------------
extern "C" void kernel_launch(void* const* d_in, const int* in_sizes, int n_in,
                              void* d_out, int out_size, void* d_ws, size_t ws_size,
                              hipStream_t stream) {
    const float* h_fea = (const float*)d_in[0];
    const float* aux   = (const float*)d_in[1];
    const float* Wq    = (const float*)d_in[2];
    const float* Wk    = (const float*)d_in[3];
    const float* Wv    = (const float*)d_in[4];
    const float* W1    = (const float*)d_in[5];
    const float* b1    = (const float*)d_in[6];
    const float* W2    = (const float*)d_in[7];
    const float* b2    = (const float*)d_in[8];
    const float* Wo    = (const float*)d_in[9];

    float* out_hwave = (float*)d_out;                               // 8*1024*512
    float* out_aux   = out_hwave + (size_t)NB * NQ * DIN;           // 8*8*1024*1024

    // workspace carve (bf16): ~34 MB total
    bf16_t* hbf = (bf16_t*)d_ws;                                    // 4,194,304
    bf16_t* Wqt = hbf + (size_t)NB * NQ * DIN;                      //   262,144
    bf16_t* Wkt = Wqt + (size_t)NH * DE * DIN;
    bf16_t* Wvt = Wkt + (size_t)NH * DE * DIN;
    bf16_t* Wot = Wvt + (size_t)NH * DE * DIN;                      //   262,144
    bf16_t* Qb  = Wot + (size_t)DIN * DIN;                          // 4,194,304
    bf16_t* Kb  = Qb  + (size_t)NH * NB * NQ * DE;
    bf16_t* Vt  = Kb  + (size_t)NH * NB * NQ * DE;

    prep_kernel<<<2048, 256, 0, stream>>>(h_fea, Wq, Wk, Wv, Wo,
                                          hbf, Wqt, Wkt, Wvt, Wot);
    qkv_kernel<<<dim3(64, NH), 256, 0, stream>>>(hbf, Wqt, Wkt, Wvt, Qb, Kb, Vt);
    attn_kernel<<<NB * (NQ / 16), 256, 0, stream>>>(Qb, Kb, Vt, aux,
                                                    W1, b1, W2, b2, Wot,
                                                    out_hwave, out_aux);
}
